// SOAP_dec_40200893890896
// MI455X (gfx1250) — matrix operand order repaired and checked by reading, not compiled, not run on hardware
//
#include <hip/hip_runtime.h>
#include <hip/hip_bf16.h>

typedef __attribute__((ext_vector_type(16))) _Float16 v16h;
typedef __attribute__((ext_vector_type(8)))  _Float16 v8h;
typedef __attribute__((ext_vector_type(8)))  float    v8f;

#define NN 4096
#define MM 4096
#define SSZ 506
#define DD 512
#define IND 512   // 3 + 3 + 506

// ---------------- WMMA helpers ----------------

__device__ __forceinline__ v16h comb(v8h x, v8h y) {
    v16h r;
#pragma unroll
    for (int i = 0; i < 8; ++i) { r[i] = x[i]; r[i + 8] = y[i]; }
    return r;
}

// 32x64 register-blocked wave GEMM: c[i][j] += A_tile(i) * B_tile(j)^T
// A, B row-major, K-contiguous. Fragment addressing per CDNA5 16-bit layouts:
//   A lane(lo,hi): row mrow+16i+lo, K chunks [hi*8,+8) and [16+hi*8,+8)
//   B lane(lo,hi): row nrow+16j+lo, K chunks [hi*8,+8) and [16+hi*8,+8)
__device__ __forceinline__ void wave_gemm_2x4(v8f (&c)[2][4],
                                              const _Float16* __restrict__ A, int lda,
                                              const _Float16* __restrict__ B, int ldb,
                                              int mrow, int nrow, int K) {
    const int lane = threadIdx.x & 31;
    const int lo = lane & 15, hi = lane >> 4;
    const _Float16* ar[2];
    const _Float16* br[4];
#pragma unroll
    for (int i = 0; i < 2; ++i) ar[i] = A + (size_t)(mrow + i * 16 + lo) * lda;
#pragma unroll
    for (int j = 0; j < 4; ++j) br[j] = B + (size_t)(nrow + j * 16 + lo) * ldb;
    for (int k0 = 0; k0 < K; k0 += 32) {
        v16h a[2], b[4];
#pragma unroll
        for (int i = 0; i < 2; ++i)
            a[i] = comb(*(const v8h*)(ar[i] + k0 + hi * 8),
                        *(const v8h*)(ar[i] + k0 + hi * 8 + 16));
#pragma unroll
        for (int j = 0; j < 4; ++j)
            b[j] = comb(*(const v8h*)(br[j] + k0 + hi * 8),
                        *(const v8h*)(br[j] + k0 + hi * 8 + 16));
#pragma unroll
        for (int i = 0; i < 2; ++i)
#pragma unroll
            for (int j = 0; j < 4; ++j)
                c[i][j] = __builtin_amdgcn_wmma_f32_16x16x32_f16(
                    false, a[i], false, b[j], (short)0, c[i][j], false, false);
    }
}

// ---------------- prep kernels ----------------

// c_k[d] = bk[d] + centre . Wk[d,3:6] + sysvec . Wk[d,6:], same for q. 512 threads total.
__global__ void k_const(const float* __restrict__ centre, const float* __restrict__ sysv,
                        const float* __restrict__ Wq, const float* __restrict__ bq,
                        const float* __restrict__ Wk, const float* __restrict__ bk,
                        float* __restrict__ cK, float* __restrict__ cQ) {
    int d = blockIdx.x * blockDim.x + threadIdx.x;           // 0..511
    const float* wk = Wk + (size_t)d * IND;
    const float* wq = Wq + (size_t)d * IND;
    float ck = bk[d], cq = bq[d];
    for (int j = 0; j < 3; ++j) { ck += centre[j] * wk[3 + j]; cq += centre[j] * wq[3 + j]; }
    for (int s = 0; s < SSZ; ++s) { ck += sysv[s] * wk[6 + s]; cq += sysv[s] * wq[6 + s]; }
    cK[d] = ck; cQ[d] = cq;
}

// Split Wv (512 x 1024 f32) into W1h/W2h (512 x 512 f16, row-major)
__global__ void k_convwv(const float* __restrict__ Wv,
                         _Float16* __restrict__ W1h, _Float16* __restrict__ W2h) {
    int i = blockIdx.x * blockDim.x + threadIdx.x;           // 0..262143
    int d = i >> 9, k = i & 511;
    W1h[i] = (_Float16)Wv[(size_t)d * 1024 + k];
    W2h[i] = (_Float16)Wv[(size_t)d * 1024 + 512 + k];
}

// keys[n,d] = nodesL[n,:3] . Wk[d,:3] + cK[d]  (f32 + f16); queries likewise (f16 only)
__global__ void k_kq(const float* __restrict__ nodesL, const float* __restrict__ nodesH,
                     const float* __restrict__ Wq, const float* __restrict__ Wk,
                     const float* __restrict__ cK, const float* __restrict__ cQ,
                     float* __restrict__ keysF, _Float16* __restrict__ keysH,
                     _Float16* __restrict__ queriesH) {
    int i = blockIdx.x * blockDim.x + threadIdx.x;           // 0 .. 4096*512-1
    int n = i >> 9, d = i & 511;
    const float* wk = Wk + (size_t)d * IND;
    const float* wq = Wq + (size_t)d * IND;
    float kf = cK[d] + nodesL[n * 3] * wk[0] + nodesL[n * 3 + 1] * wk[1] + nodesL[n * 3 + 2] * wk[2];
    float qf = cQ[d] + nodesH[n * 3] * wq[0] + nodesH[n * 3 + 1] * wq[1] + nodesH[n * 3 + 2] * wq[2];
    keysF[i] = kf;
    keysH[i] = (_Float16)kf;
    queriesH[i] = (_Float16)qf;
}

// ---------------- WMMA GEMM kernels ----------------

// VqT[d, j] = (queries @ W1^T + bv)[j, d], stored f16 (512 x 4096 row-major)
// 1024 waves: 128 m-tiles (32 rows) x 8 n-blocks (64 cols of D)
__global__ void k_gemm_vqt(const _Float16* __restrict__ Q, const _Float16* __restrict__ W1,
                           const float* __restrict__ bv, _Float16* __restrict__ VqT) {
    int wave = blockIdx.x * (blockDim.x >> 5) + (threadIdx.x >> 5);
    int mrow = (wave >> 3) * 32;
    int nrow = (wave & 7) * 64;
    v8f c[2][4] = {};
    wave_gemm_2x4(c, Q, DD, W1, DD, mrow, nrow, DD);
    const int lane = threadIdx.x & 31, lo = lane & 15, hi = lane >> 4;
    // C/D layout: VGPR r = element (M = r + 8*hi, N = lo). Transposed store:
    // VqT row = n (bias index), 8 M-contiguous halves -> one b128 store.
#pragma unroll
    for (int j = 0; j < 4; ++j) {
        float bias = bv[nrow + j * 16 + lo];
#pragma unroll
        for (int i = 0; i < 2; ++i) {
            v8h t;
#pragma unroll
            for (int r = 0; r < 8; ++r) t[r] = (_Float16)(c[i][j][r] + bias);
            *(v8h*)(VqT + (size_t)(nrow + j * 16 + lo) * MM + mrow + i * 16 + 8 * hi) = t;
        }
    }
}

// S = keys @ queries^T (f16, 4096 x 4096). 8192 waves: 128 x 64 tiles of 32x64.
__global__ void k_gemm_logits(const _Float16* __restrict__ Kh, const _Float16* __restrict__ Qh,
                              _Float16* __restrict__ S) {
    int wave = blockIdx.x * (blockDim.x >> 5) + (threadIdx.x >> 5);
    int mrow = (wave >> 6) * 32;
    int nrow = (wave & 63) * 64;
    v8f c[2][4] = {};
    wave_gemm_2x4(c, Kh, DD, Qh, DD, mrow, nrow, DD);
    const int lane = threadIdx.x & 31, lo = lane & 15, hi = lane >> 4;
#pragma unroll
    for (int i = 0; i < 2; ++i)
#pragma unroll
        for (int r = 0; r < 8; ++r) {
            size_t row = (size_t)(mrow + i * 16 + r + 8 * hi) * MM;
#pragma unroll
            for (int j = 0; j < 4; ++j)
                S[row + nrow + j * 16 + lo] = (_Float16)c[i][j][r];
        }
}

// In-place rowwise softmax over S (f16 storage, f32 math). One block per row.
__global__ void k_softmax(_Float16* __restrict__ S) {
    __shared__ float red[256];
    _Float16* p = S + (size_t)blockIdx.x * MM;
    float v[16];
    float mx = -1e30f;
#pragma unroll
    for (int i = 0; i < 16; ++i) { v[i] = (float)p[threadIdx.x + i * 256]; mx = fmaxf(mx, v[i]); }
    red[threadIdx.x] = mx; __syncthreads();
    for (int s = 128; s > 0; s >>= 1) {
        if (threadIdx.x < s) red[threadIdx.x] = fmaxf(red[threadIdx.x], red[threadIdx.x + s]);
        __syncthreads();
    }
    mx = red[0]; __syncthreads();
    float sum = 0.f;
#pragma unroll
    for (int i = 0; i < 16; ++i) { v[i] = __expf(v[i] - mx); sum += v[i]; }
    red[threadIdx.x] = sum; __syncthreads();
    for (int s = 128; s > 0; s >>= 1) {
        if (threadIdx.x < s) red[threadIdx.x] += red[threadIdx.x + s];
        __syncthreads();
    }
    float inv = 1.0f / red[0];
#pragma unroll
    for (int i = 0; i < 16; ++i) p[threadIdx.x + i * 256] = (_Float16)(v[i] * inv);
}

// out = keysF + keys @ W2^T + att @ Vq   (att@Vq done as att @ VqT^T)
// 1024 waves: 128 m-tiles x 8 n-blocks of 64 cols over D.
__global__ void k_gemm_out(const _Float16* __restrict__ Kh, const _Float16* __restrict__ W2,
                           const _Float16* __restrict__ S, const _Float16* __restrict__ VqT,
                           const float* __restrict__ keysF, float* __restrict__ out) {
    int wave = blockIdx.x * (blockDim.x >> 5) + (threadIdx.x >> 5);
    int mrow = (wave >> 3) * 32;
    int nrow = (wave & 7) * 64;
    v8f c[2][4] = {};
    wave_gemm_2x4(c, Kh, DD, W2, DD, mrow, nrow, DD);     // K = 512
    wave_gemm_2x4(c, S, MM, VqT, MM, mrow, nrow, MM);     // K = 4096
    const int lane = threadIdx.x & 31, lo = lane & 15, hi = lane >> 4;
#pragma unroll
    for (int i = 0; i < 2; ++i)
#pragma unroll
        for (int r = 0; r < 8; ++r) {
            size_t row = (size_t)(mrow + i * 16 + r + 8 * hi) * DD;
#pragma unroll
            for (int j = 0; j < 4; ++j) {
                size_t idx = row + nrow + j * 16 + lo;
                out[idx] = c[i][j][r] + keysF[idx];
            }
        }
}

// ---------------- launch ----------------

extern "C" void kernel_launch(void* const* d_in, const int* in_sizes, int n_in,
                              void* d_out, int out_size, void* d_ws, size_t ws_size,
                              hipStream_t stream) {
    const float* nodesL = (const float*)d_in[0];
    const float* nodesH = (const float*)d_in[1];
    const float* centre = (const float*)d_in[2];
    const float* sysv   = (const float*)d_in[3];
    const float* Wq     = (const float*)d_in[4];
    const float* bq     = (const float*)d_in[5];
    const float* Wk     = (const float*)d_in[6];
    const float* bk     = (const float*)d_in[7];
    const float* Wv     = (const float*)d_in[8];
    const float* bv     = (const float*)d_in[9];
    float* out = (float*)d_out;

    char* w = (char*)d_ws;
    float*    cK       = (float*)(w + 0);                 //   2 KB
    float*    cQ       = (float*)(w + 2048);              //   2 KB
    _Float16* W1h      = (_Float16*)(w + 4096);           // 512 KB
    _Float16* W2h      = (_Float16*)(w + 528384);         // 512 KB
    _Float16* keysH    = (_Float16*)(w + 1052672);        //   4 MB
    _Float16* queriesH = (_Float16*)(w + 5246976);        //   4 MB
    _Float16* VqT      = (_Float16*)(w + 9441280);        //   4 MB
    float*    keysF    = (float*)(w + 13635584);          //   8 MB
    _Float16* S        = (_Float16*)(w + 22024192);       //  32 MB  (logits, then att in-place)

    k_const      <<<2,    256, 0, stream>>>(centre, sysv, Wq, bq, Wk, bk, cK, cQ);
    k_convwv     <<<1024, 256, 0, stream>>>(Wv, W1h, W2h);
    k_kq         <<<8192, 256, 0, stream>>>(nodesL, nodesH, Wq, Wk, cK, cQ, keysF, keysH, queriesH);
    k_gemm_vqt   <<<256,  128, 0, stream>>>(queriesH, W1h, bv, VqT);
    k_gemm_logits<<<2048, 128, 0, stream>>>(keysH, queriesH, S);
    k_softmax    <<<4096, 256, 0, stream>>>(S);
    k_gemm_out   <<<256,  128, 0, stream>>>(keysH, W2h, S, VqT, keysF, out);
}
